// BaseEncoderBiRNN_42889543417937
// MI455X (gfx1250) — matrix operand order repaired and checked by reading, not compiled, not run on hardware
//
#include <hip/hip_runtime.h>
#include <hip/hip_bf16.h>
#include <math.h>

// Problem dims (compile-time constants from the reference)
#define TT 2048
#define II 1024
#define HH 1024
#define GG 4096   // 4*H

typedef __attribute__((ext_vector_type(16))) __bf16 v16bf;
typedef __attribute__((ext_vector_type(8)))  __bf16 v8bf;
typedef __attribute__((ext_vector_type(8)))  float  v8f;
typedef __attribute__((ext_vector_type(4)))  unsigned v4u;
typedef __attribute__((ext_vector_type(8)))  int      v8i;
typedef __attribute__((ext_vector_type(4)))  int      v4i;

// ---------------------------------------------------------------------------
// fp32 -> bf16 conversion (grid-stride)
// ---------------------------------------------------------------------------
__global__ __launch_bounds__(256) void cvt_bf16_kernel(const float* __restrict__ src,
                                                       __bf16* __restrict__ dst, int n) {
    int i = blockIdx.x * blockDim.x + threadIdx.x;
    int stride = gridDim.x * blockDim.x;
    for (; i < n; i += stride) dst[i] = (__bf16)src[i];
}

// ---------------------------------------------------------------------------
// zero h double-buffers and grid-barrier counters (ws is poisoned by harness)
// ---------------------------------------------------------------------------
__global__ __launch_bounds__(256) void init_ws_kernel(float* __restrict__ hbuf,
                                                      unsigned* __restrict__ bar) {
    int i = blockIdx.x * blockDim.x + threadIdx.x;
    if (i < 4 * HH) hbuf[i] = 0.0f;   // 2 dirs x 2 parity x H
    if (i < 4)      bar[i]  = 0u;     // 2 dirs x {count, gen}
}

// ---------------------------------------------------------------------------
// GEMM: xg[t][g] = sum_k x[t][k]*Wih[g][k] + (bih[g]+bhh[g])
// WMMA bf16 16x16x32, fp32 accum. Each wave: 16M x 64N; WG (8 waves): 32M x 256N.
// A tile (32x32 bf16) staged into LDS by the Tensor Data Mover (one wave issues
// tensor_load_to_lds with an ISA-8 D# descriptor, then s_wait_tensorcnt).
// B fragments come straight from global via plain global_load_b128 (Wih rows are
// k-contiguous: each lane reads its B column directly), eliminating LDS transpose.
// ---------------------------------------------------------------------------
__global__ __launch_bounds__(256) void gemm_xg_kernel(const __bf16* __restrict__ xb,
                                                      const __bf16* __restrict__ wih,
                                                      const float* __restrict__ bih,
                                                      const float* __restrict__ bhh,
                                                      float* __restrict__ xg) {
    __shared__ __attribute__((aligned(16))) __bf16 ldsA[32][32];

    const int tid   = threadIdx.x;
    const int lane  = tid & 31;
    const int wave  = tid >> 5;   // 0..7
    const int waveM = wave >> 2;  // 0..1
    const int waveN = wave & 3;   // 0..3

    const int tBase = blockIdx.y * 32;   // M (time rows)
    const int nBase = blockIdx.x * 256;  // N (gate cols)

    // LDS byte address of the A staging tile (generic ptr low 32 bits = LDS offset)
    const unsigned ldsAaddr = (unsigned)(size_t)(void*)&ldsA[0][0];

    v8f acc[4] = {};

    for (int kb = 0; kb < II; kb += 32) {
        __syncthreads();

        // ---- TDM: DMA the 32x32 bf16 A tile from x_bf16 into LDS -------------
        if (wave == 0) {
            const unsigned long long ga =
                (unsigned long long)(size_t)(xb + (size_t)tBase * II + kb);
            // D# group0: count=1 | lds_addr | global_addr[56:0] | type=2
            v4u g0 = { 1u,
                       ldsAaddr,
                       (unsigned)(ga & 0xffffffffull),
                       (unsigned)((ga >> 32) & 0x01ffffffull) | (2u << 30) };
            // D# group1: wg_mask=0, data_size=1(2B); tensor_dim0=1024;
            // tensor_dim1=2048; tile_dim0=32; tile_dim1=32; dim0_stride=1024
            v8i g1 = { 0x00010000,            // data_size=2B in bits[17:16]
                       (int)(1024u << 16),    // tensor_dim0[15:0] in [31:16]
                       (int)(2048u << 16),    // tensor_dim1[15:0] in bits[95:80]
                       (int)(32u << 16),      // tile_dim0 in bits[127:112]
                       32,                    // tile_dim1 in bits[143:128]
                       1024,                  // tensor_dim0_stride[31:0]
                       0, 0 };
            v4i gz = { 0, 0, 0, 0 };
#if defined(__clang_major__) && (__clang_major__ >= 23)
            v8i g4z = { 0, 0, 0, 0, 0, 0, 0, 0 };
            __builtin_amdgcn_tensor_load_to_lds(g0, g1, gz, gz, g4z, 0);
#else
            __builtin_amdgcn_tensor_load_to_lds(g0, g1, gz, gz, 0);
#endif
            __builtin_amdgcn_s_wait_tensorcnt(0);
        }
        __syncthreads();

        // ---- A fragment (16x32 bf16: lane<16 holds K 0..7,16..23; lane>=16 K 8..15,24..31)
        const int hi   = lane >> 4;          // 0|1
        const int aRow = waveM * 16 + (lane & 15);
        v8bf a_lo = *(const v8bf*)&ldsA[aRow][hi * 8];
        v8bf a_hi = *(const v8bf*)&ldsA[aRow][16 + hi * 8];
        v16bf afrag = __builtin_shufflevector(a_lo, a_hi,
                         0,1,2,3,4,5,6,7,8,9,10,11,12,13,14,15);

        // ---- B fragments via plain b128 loads straight from global ----------
#pragma unroll
        for (int ni = 0; ni < 4; ++ni) {
            const int n = nBase + waveN * 64 + ni * 16 + (lane & 15);
            // lane addr: row n of Wih, 8-half chunk selected by lane group
            const unsigned long long a0 =
                (unsigned long long)(size_t)(wih + (size_t)n * II + kb + 8 * hi);
            const unsigned long long a1 = a0 + 32ull;   // K+16 chunk (16 halfs)
            v4u b0, b1;
            asm volatile("global_load_b128 %0, %2, off\n\t"
                         "global_load_b128 %1, %3, off\n\t"
                         "s_wait_loadcnt 0x0"
                         : "=&v"(b0), "=&v"(b1)
                         : "v"(a0), "v"(a1)
                         : "memory");
            union { v4u q[2]; v16bf v; } bu;
            bu.q[0] = b0; bu.q[1] = b1;
            acc[ni] = __builtin_amdgcn_wmma_f32_16x16x32_bf16(
                false, afrag, false, bu.v, (short)0, acc[ni], false, false);
        }
        // pull the next K-slab of Wih toward the caches
        __builtin_prefetch(wih + (size_t)(nBase + waveN * 64 + lane) * II + kb + 32, 0, 1);
    }

    // ---- epilogue: bias add + store (C layout: VGPR r -> row r+8*hi, col lane&15)
    const int hi = lane >> 4;
#pragma unroll
    for (int ni = 0; ni < 4; ++ni) {
        const int gcol = nBase + waveN * 64 + ni * 16 + (lane & 15);
        const float bias = bih[gcol] + bhh[gcol];
#pragma unroll
        for (int r = 0; r < 8; ++r) {
            const int t = tBase + waveM * 16 + r + 8 * hi;
            xg[(size_t)t * GG + gcol] = acc[ni][r] + bias;
        }
    }
}

// ---------------------------------------------------------------------------
// Recurrence: persistent kernel, 64 WGs per direction (128 total), each WG
// owns 16 h-elements (=> 64 gate rows of Whh, L2-resident). Double-buffered h
// in global memory; per-step grid barrier via atomic count/generation.
// ---------------------------------------------------------------------------
#define WGS_PER_DIR 64

__global__ __launch_bounds__(256) void lstm_rec_kernel(const float* __restrict__ WhhF,
                                                       const float* __restrict__ WhhR,
                                                       const float* __restrict__ xgF,
                                                       const float* __restrict__ xgR,
                                                       float* __restrict__ out,
                                                       float* __restrict__ hbuf,
                                                       unsigned* __restrict__ bar) {
    const int dir   = blockIdx.x / WGS_PER_DIR;   // 0 fwd, 1 rev
    const int slice = blockIdx.x % WGS_PER_DIR;   // 0..63
    const int j0    = slice * 16;

    const float* Whh = dir ? WhhR : WhhF;
    const float* xg  = dir ? xgR  : xgF;
    float*    hcur = hbuf + dir * (2 * HH);       // 2 parities x H
    unsigned* cnt  = bar + dir * 2;
    unsigned* gen  = bar + dir * 2 + 1;

    __shared__ __attribute__((aligned(16))) float h_lds[HH];
    __shared__ float gates[64];
    __shared__ float c_lds[16];

    const int tid  = threadIdx.x;
    const int lane = tid & 31;
    const int wave = tid >> 5;

    if (tid < 16) c_lds[tid] = 0.0f;

    for (int s = 0; s < TT; ++s) {
        __syncthreads();
        // stage previous h into LDS (buf parity s&1; buf[0] zeroed by init)
        const float* hin = hcur + (s & 1) * HH;
        *(float4*)&h_lds[tid * 4] = *(const float4*)&hin[tid * 4];
        __syncthreads();

        // 8 gate rows per wave; 32-lane dot of length 1024, shuffle-reduce
#pragma unroll
        for (int rr = 0; rr < 8; ++rr) {
            const int idx = wave * 8 + rr;   // 0..63
            const int gi  = idx >> 4;        // gate i/f/g/o
            const int jj  = idx & 15;
            const float* wrow = Whh + (size_t)(gi * HH + j0 + jj) * HH;
            float sum = 0.0f;
#pragma unroll
            for (int c = 0; c < 8; ++c) {
                const int k = lane * 4 + c * 128;
                float4 w  = *(const float4*)&wrow[k];
                float4 hv = *(const float4*)&h_lds[k];
                sum += w.x * hv.x + w.y * hv.y + w.z * hv.z + w.w * hv.w;
            }
#pragma unroll
            for (int off = 16; off > 0; off >>= 1)
                sum += __shfl_xor(sum, off, 32);
            if (lane == 0) gates[idx] = sum;
        }
        __syncthreads();

        if (tid < 16) {
            const int jj   = tid;
            const int trow = dir ? (TT - 1 - s) : s;  // rev consumes x[T-1-s]
            const float* xr = xg + (size_t)trow * GG;
            float ip = gates[jj]      + xr[          j0 + jj];
            float fp = gates[16 + jj] + xr[HH      + j0 + jj];
            float gp = gates[32 + jj] + xr[2 * HH  + j0 + jj];
            float op = gates[48 + jj] + xr[3 * HH  + j0 + jj];
            float ig = 1.0f / (1.0f + __expf(-ip));
            float fg = 1.0f / (1.0f + __expf(-fp));
            float gv = tanhf(gp);
            float og = 1.0f / (1.0f + __expf(-op));
            float c  = fg * c_lds[jj] + ig * gv;
            c_lds[jj] = c;
            float h  = og * tanhf(c);
            hcur[((s + 1) & 1) * HH + j0 + jj] = h;
            out[(size_t)trow * (2 * HH) + dir * HH + j0 + jj] = h;
            if (s == TT - 1) {
                out[(size_t)TT * 2 * HH + dir * HH + j0 + jj] = h;            // hidden
                out[(size_t)TT * 2 * HH + 2 * HH + dir * HH + j0 + jj] = c;   // cell
            }
        }

        // per-direction grid barrier (generation counter)
        __threadfence();
        __syncthreads();
        if (tid == 0) {
            unsigned arrived = atomicAdd(cnt, 1u);
            if (arrived == WGS_PER_DIR - 1) {
                atomicExch(cnt, 0u);
                __threadfence();
                atomicAdd(gen, 1u);
            }
            const unsigned target = (unsigned)(s + 1);
            while (__hip_atomic_load(gen, __ATOMIC_ACQUIRE, __HIP_MEMORY_SCOPE_AGENT) < target) {
                __builtin_amdgcn_s_sleep(2);
            }
        }
        __syncthreads();
    }
}

// ---------------------------------------------------------------------------
// Launcher. Workspace layout (bytes):
//   [0,4M)    x bf16          [4M,12M)  fwd_Wih bf16   [12M,20M) rev_Wih bf16
//   [20M,52M) xg_fwd f32      [52M,84M) xg_rev f32
//   [84M, +16K) h double buffers   [+64K) barrier counters
// Total ~84.1 MB of d_ws.
// ---------------------------------------------------------------------------
extern "C" void kernel_launch(void* const* d_in, const int* in_sizes, int n_in,
                              void* d_out, int out_size, void* d_ws, size_t ws_size,
                              hipStream_t stream) {
    const float* x    = (const float*)d_in[0];
    const float* fWih = (const float*)d_in[1];
    const float* fWhh = (const float*)d_in[2];
    const float* fbih = (const float*)d_in[3];
    const float* fbhh = (const float*)d_in[4];
    const float* rWih = (const float*)d_in[5];
    const float* rWhh = (const float*)d_in[6];
    const float* rbih = (const float*)d_in[7];
    const float* rbhh = (const float*)d_in[8];
    float* out = (float*)d_out;

    char* ws = (char*)d_ws;
    __bf16*   xb   = (__bf16*)(ws);
    __bf16*   wfb  = (__bf16*)(ws + (size_t)(4u  << 20));
    __bf16*   wrb  = (__bf16*)(ws + (size_t)(12u << 20));
    float*    xgF  = (float*) (ws + (size_t)(20u << 20));
    float*    xgR  = (float*) (ws + (size_t)(52u << 20));
    float*    hbuf = (float*) (ws + (size_t)(84u << 20));
    unsigned* bar  = (unsigned*)(ws + (size_t)(84u << 20) + (1u << 16));

    init_ws_kernel<<<16, 256, 0, stream>>>(hbuf, bar);

    cvt_bf16_kernel<<<1024, 256, 0, stream>>>(x,    xb,  TT * II);
    cvt_bf16_kernel<<<1024, 256, 0, stream>>>(fWih, wfb, GG * II);
    cvt_bf16_kernel<<<1024, 256, 0, stream>>>(rWih, wrb, GG * II);

    dim3 ggrid(GG / 256, TT / 32);
    gemm_xg_kernel<<<ggrid, 256, 0, stream>>>(xb, wfb, fbih, fbhh, xgF);
    gemm_xg_kernel<<<ggrid, 256, 0, stream>>>(xb, wrb, rbih, rbhh, xgR);

    lstm_rec_kernel<<<2 * WGS_PER_DIR, 256, 0, stream>>>(fWhh, rWhh, xgF, xgR,
                                                         out, hbuf, bar);
}
